// GraphAttentionLayer_33217277067887
// MI455X (gfx1250) — matrix operand order repaired and checked by reading, not compiled, not run on hardware
//
#include <hip/hip_runtime.h>

#define NNODES 8192
#define NHEADS 4
#define NHID 64

typedef __attribute__((ext_vector_type(4)))  float        f4;
typedef __attribute__((ext_vector_type(8)))  float        v8f;
typedef __attribute__((ext_vector_type(8)))  unsigned int v8u;
typedef __attribute__((ext_vector_type(16))) __bf16       v16bf;

__device__ __forceinline__ unsigned short f2bf(float x) {
    unsigned u = __float_as_uint(x);
    u += 0x7FFFu + ((u >> 16) & 1u);   // round-to-nearest-even
    return (unsigned short)(u >> 16);
}

__device__ __forceinline__ float mkem(float srow, int aij, float t) {
    float e = srow + t;
    e = fmaxf(e, 0.2f * e);            // leaky relu
    return aij > 0 ? e : -3.0e38f;     // mask -> huge negative (exp underflows to 0)
}

// ---------------------------------------------------------------------------
// Kernel 1: projection  ht[h][n][d] = (h @ W), plus bf16 transposed copy
//           vt[h][d][n] for WMMA B-matrix consumption.
// Block: 256 threads, 16 nodes per block. hsT staged transposed in LDS.
// ---------------------------------------------------------------------------
__global__ __launch_bounds__(256) void k_proj(const float* __restrict__ hin,
                                              const float* __restrict__ Wg,
                                              float* __restrict__ ht,
                                              unsigned short* __restrict__ vt) {
    __shared__ float hsT[256 * 16];          // [k][n], 16 KB
    const int t  = threadIdx.x;
    const int n0 = blockIdx.x * 16;

    {   // load h tile [16 x 256] transposed into LDS
        const int n  = t >> 4;               // 0..15
        const int k0 = (t & 15) << 4;        // 0,16,...,240
        const float* hp = hin + (size_t)(n0 + n) * 256 + k0;
        #pragma unroll
        for (int q = 0; q < 4; ++q) {
            f4 v = ((const f4*)hp)[q];
            hsT[(k0 + 4 * q + 0) * 16 + n] = v.x;
            hsT[(k0 + 4 * q + 1) * 16 + n] = v.y;
            hsT[(k0 + 4 * q + 2) * 16 + n] = v.z;
            hsT[(k0 + 4 * q + 3) * 16 + n] = v.w;
        }
    }
    __syncthreads();

    const int c = t;                          // output column 0..255
    f4 acc[4] = {{0,0,0,0},{0,0,0,0},{0,0,0,0},{0,0,0,0}};
    for (int k = 0; k < 256; ++k) {
        float wv = Wg[k * 256 + c];           // coalesced across threads
        const f4* rr = (const f4*)&hsT[k * 16];   // LDS broadcast
        #pragma unroll
        for (int q = 0; q < 4; ++q) acc[q] += rr[q] * wv;
    }

    const int head = c >> 6, d = c & 63;
    float av[16];
    #pragma unroll
    for (int q = 0; q < 4; ++q) {
        av[4*q+0] = acc[q].x; av[4*q+1] = acc[q].y;
        av[4*q+2] = acc[q].z; av[4*q+3] = acc[q].w;
    }
    #pragma unroll
    for (int n = 0; n < 16; ++n)
        ht[((size_t)head * NNODES + n0 + n) * NHID + d] = av[n];

    v8u pk;
    #pragma unroll
    for (int q = 0; q < 8; ++q)
        pk[q] = (unsigned)f2bf(av[2*q]) | ((unsigned)f2bf(av[2*q+1]) << 16);
    *(v8u*)(vt + ((size_t)head * NHID + d) * NNODES + n0) = pk;
}

// ---------------------------------------------------------------------------
// Kernel 2: src[h][n] = ht[h][n]·a[h][:64],  tgt[h][n] = ht[h][n]·a[h][64:]
// ---------------------------------------------------------------------------
__global__ __launch_bounds__(256) void k_srctgt(const float* __restrict__ ht,
                                                const float* __restrict__ av,
                                                float* __restrict__ srcv,
                                                float* __restrict__ tgtv) {
    const int idx  = blockIdx.x * 256 + threadIdx.x;   // head*8192 + n
    const int head = idx >> 13;
    const float* row = ht + (size_t)idx * NHID;
    const float* as  = av + head * 128;
    const float* at  = as + 64;
    float s = 0.f, tt = 0.f;
    #pragma unroll
    for (int d = 0; d < 64; d += 4) {
        f4 r = *(const f4*)(row + d);
        f4 x = *(const f4*)(as + d);
        f4 y = *(const f4*)(at + d);
        s  += r.x * x.x + r.y * x.y + r.z * x.z + r.w * x.w;
        tt += r.x * y.x + r.y * y.y + r.z * y.z + r.w * y.w;
    }
    srcv[idx] = s;
    tgtv[idx] = tt;
}

// ---------------------------------------------------------------------------
// Kernel 3: fused masked-softmax attention with online softmax + bf16 WMMA.
// Block = 128 threads = 4 waves; wave w handles head w for 16 query rows.
// Adjacency read once per row-block; all heads share it through cache.
// ---------------------------------------------------------------------------
__global__ __launch_bounds__(128) void k_attn(const int* __restrict__ adj,
                                              const float* __restrict__ srcv,
                                              const float* __restrict__ tgtv,
                                              const unsigned short* __restrict__ vt,
                                              float* __restrict__ out) {
    __shared__ float red[NHEADS][16][NHID];    // 16 KB, per-head tiles (det. reduce)

    const int i0   = blockIdx.x * 16;
    const int w    = threadIdx.x >> 5;         // head
    const unsigned lane = threadIdx.x & 31u;
    const int r    = lane & 15;                // A-matrix row within tile
    const int half = lane >> 4;                // 0: K 0..7/16..23, 1: +8
    const int kb   = half * 8;

    const float  srow = srcv[(size_t)w * NNODES + i0 + r];
    const float* tgh  = tgtv + (size_t)w * NNODES;
    const int*   arow = adj + (size_t)(i0 + r) * NNODES;
    const unsigned short* vth = vt + (size_t)w * NHID * NNODES;

    v8f acc[4];
    #pragma unroll
    for (int cB = 0; cB < 4; ++cB) acc[cB] = (v8f){0,0,0,0,0,0,0,0};
    float m = -3.0e38f, l = 0.f;

    for (int j0 = 0; j0 < NNODES; j0 += 32) {
        __builtin_prefetch(arow + j0 + kb + 128, 0, 1);   // global_prefetch ahead

        const int4* ap = (const int4*)(arow + j0 + kb);
        const int4  A0 = ap[0], A1 = ap[1], A2 = ap[4], A3 = ap[5];
        const f4*   tp = (const f4*)(tgh + j0 + kb);
        const f4    T0 = tp[0], T1 = tp[1], T2 = tp[4], T3 = tp[5];

        float em[16];
        em[ 0]=mkem(srow,A0.x,T0.x); em[ 1]=mkem(srow,A0.y,T0.y);
        em[ 2]=mkem(srow,A0.z,T0.z); em[ 3]=mkem(srow,A0.w,T0.w);
        em[ 4]=mkem(srow,A1.x,T1.x); em[ 5]=mkem(srow,A1.y,T1.y);
        em[ 6]=mkem(srow,A1.z,T1.z); em[ 7]=mkem(srow,A1.w,T1.w);
        em[ 8]=mkem(srow,A2.x,T2.x); em[ 9]=mkem(srow,A2.y,T2.y);
        em[10]=mkem(srow,A2.z,T2.z); em[11]=mkem(srow,A2.w,T2.w);
        em[12]=mkem(srow,A3.x,T3.x); em[13]=mkem(srow,A3.y,T3.y);
        em[14]=mkem(srow,A3.z,T3.z); em[15]=mkem(srow,A3.w,T3.w);

        float cm = em[0];
        #pragma unroll
        for (int i = 1; i < 16; ++i) cm = fmaxf(cm, em[i]);
        cm = fmaxf(cm, __shfl_xor(cm, 16));           // row max across lane pair
        const float mnew = fmaxf(m, cm);
        const float f    = __expf(m - mnew);          // ==1 when no update

        if (__any((int)(mnew > m))) {                 // wave-uniform rescale
            #pragma unroll
            for (int v = 0; v < 8; ++v) {
                float fv = __shfl(f, half * 8 + v);   // factor for C/D row 8*half+v
                #pragma unroll
                for (int cB = 0; cB < 4; ++cB) acc[cB][v] *= fv;
            }
        }

        float p[16], ps = 0.f;
        #pragma unroll
        for (int i = 0; i < 16; ++i) { p[i] = __expf(em[i] - mnew); ps += p[i]; }
        l = fmaf(l, f, ps);
        m = mnew;

        v8u pbits;                                    // pack P -> bf16 A-matrix
        #pragma unroll
        for (int q = 0; q < 8; ++q)
            pbits[q] = (unsigned)f2bf(p[2*q]) | ((unsigned)f2bf(p[2*q+1]) << 16);
        const v16bf pa = __builtin_bit_cast(v16bf, pbits);

        const unsigned short* vb = vth + j0 + half * 8;
        #pragma unroll
        for (int cB = 0; cB < 4; ++cB) {
            const uint4* bp = (const uint4*)(vb + (size_t)(cB * 16 + r) * NNODES);
            const uint4 b0 = bp[0], b1 = bp[2];
            v8u bv = {b0.x, b0.y, b0.z, b0.w, b1.x, b1.y, b1.z, b1.w};
            v16bf pb = __builtin_bit_cast(v16bf, bv);
            acc[cB] = __builtin_amdgcn_wmma_f32_16x16x32_bf16(
                          false, pa, false, pb, (short)0, acc[cB], false, false);
        }
    }

    // finalize: merge lane-pair partial sums, scale by 0.25/l, stage to LDS
    const float lt = l + __shfl_xor(l, 16);
    #pragma unroll
    for (int v = 0; v < 8; ++v) {
        float lv  = __shfl(lt, half * 8 + v);
        float inv = 0.25f / lv;
        const int row = half * 8 + v;
        #pragma unroll
        for (int cB = 0; cB < 4; ++cB)
            red[w][row][cB * 16 + r] = acc[cB][v] * inv;
    }
    __syncthreads();

    // deterministic head reduction + store
    {
        const int t   = threadIdx.x;      // 0..127
        const int row = t >> 3;           // 0..15
        const int c0  = (t & 7) * 8;
        f4 s0 = {0,0,0,0}, s1 = {0,0,0,0};
        #pragma unroll
        for (int hh = 0; hh < NHEADS; ++hh) {
            const f4* pr = (const f4*)&red[hh][row][c0];
            s0 += pr[0]; s1 += pr[1];
        }
        f4* op = (f4*)(out + (size_t)(i0 + row) * NHID + c0);
        op[0] = s0; op[1] = s1;
    }
}

// ---------------------------------------------------------------------------
extern "C" void kernel_launch(void* const* d_in, const int* in_sizes, int n_in,
                              void* d_out, int out_size, void* d_ws, size_t ws_size,
                              hipStream_t stream) {
    (void)in_sizes; (void)n_in; (void)out_size; (void)ws_size;
    const float* h   = (const float*)d_in[0];
    const int*   adj = (const int*)d_in[1];
    const float* W   = (const float*)d_in[2];
    const float* a   = (const float*)d_in[3];
    float* out = (float*)d_out;

    float* ht   = (float*)d_ws;                       // [4][8192][64] f32 (8 MB)
    float* srcv = ht + (size_t)NHEADS * NNODES * NHID;  // [4][8192]
    float* tgtv = srcv + (size_t)NHEADS * NNODES;       // [4][8192]
    unsigned short* vt = (unsigned short*)(tgtv + (size_t)NHEADS * NNODES); // [4][64][8192] bf16 (4 MB)

    k_proj  <<<NNODES / 16, 256, 0, stream>>>(h, W, ht, vt);
    k_srctgt<<<(NHEADS * NNODES) / 256, 256, 0, stream>>>(ht, a, srcv, tgtv);
    k_attn  <<<NNODES / 16, 128, 0, stream>>>(adj, srcv, tgtv, vt, out);
}
